// NLB_Head_77481210020379
// MI455X (gfx1250) — matrix operand order repaired and checked by reading, not compiled, not run on hardware
//
#include <hip/hip_runtime.h>
#include <hip/hip_bf16.h>
#include <math.h>

#define C_DIM  1536
#define L_DIM  1024
#define B_DIM  32
#define D2_DIM 3072

typedef __attribute__((ext_vector_type(16))) __bf16 v16bf;
typedef __attribute__((ext_vector_type(8)))  float  v8f;

// ---------------------------------------------------------------------------
// Skinny GEMM: OUT[32 x N] = act(A[32 x K] @ op(W) + bias (+ residual))
// One wave computes BOTH 16x16 m-tiles of a 16-wide column strip, so the
// B (weight) fragment is loaded once and reused by two back-to-back
// v_wmma_f32_16x16x32_bf16 ops (independent accumulators -> co-executable).
// fp32 memory -> bf16 fragments, fp32 accumulation.
// A fragment layout (16x32 bf16): lane L -> m=L&15, kh=L>>4;
//   elems 0..7  -> K = kh*8 + e        (contiguous -> b128 loads)
//   elems 8..15 -> K = 16 + kh*8 + e-8 (contiguous)
// B fragment layout (32x16 bf16): lane L -> n=L&15, kh=L>>4; elem e -> same K as A.
// ---------------------------------------------------------------------------
template<bool TRANS_W, bool RELU, bool HAS_BIAS, bool HAS_RES>
__global__ void gemm32_bf16_wmma(const float* __restrict__ A, int K,
                                 const float* __restrict__ W,
                                 const float* __restrict__ bias,
                                 const float* __restrict__ residual, int res_ld,
                                 float* __restrict__ OUT, int N, int out_ld,
                                 int out_coloff)
{
    const int lane  = threadIdx.x;     // wave32
    const int ntile = blockIdx.x;      // N/16 tiles
    const int m  = lane & 15;
    const int kh = lane >> 4;

    v8f acc0 = {};
    v8f acc1 = {};
    const float* __restrict__ arow0 = A + (size_t)m * K;          // m-tile 0
    const float* __restrict__ arow1 = A + (size_t)(16 + m) * K;   // m-tile 1

    for (int kb = 0; kb < K; kb += 32) {
        v16bf a0, a1, wf;
        #pragma unroll
        for (int e = 0; e < 8; ++e) {
            a0[e]     = (__bf16)arow0[kb + kh * 8 + e];
            a0[e + 8] = (__bf16)arow0[kb + 16 + kh * 8 + e];
            a1[e]     = (__bf16)arow1[kb + kh * 8 + e];
            a1[e + 8] = (__bf16)arow1[kb + 16 + kh * 8 + e];
        }
        const int nc = ntile * 16 + (lane & 15);
        #pragma unroll
        for (int e = 0; e < 16; ++e) {
            const int k = kb + (e < 8 ? kh * 8 + e : 16 + kh * 8 + (e - 8));
            float w = TRANS_W ? W[(size_t)nc * K + k]
                              : W[(size_t)k * N + nc];
            wf[e] = (__bf16)w;
        }
        acc0 = __builtin_amdgcn_wmma_f32_16x16x32_bf16(
                   false, a0, false, wf, (short)0, acc0, false, false);
        acc1 = __builtin_amdgcn_wmma_f32_16x16x32_bf16(
                   false, a1, false, wf, (short)0, acc1, false, false);
    }

    const int n    = lane & 15;
    const int half = lane >> 4;
    const int cg   = ntile * 16 + n;
    const float bv = HAS_BIAS ? bias[cg] : 0.f;

    #pragma unroll
    for (int r = 0; r < 8; ++r) {
        const int row0 = r + 8 * half;          // m-tile 0 row
        const int row1 = 16 + r + 8 * half;     // m-tile 1 row
        float v0 = acc0[r] + bv;
        float v1 = acc1[r] + bv;
        if (HAS_RES) {
            v0 += residual[(size_t)row0 * res_ld + cg];
            v1 += residual[(size_t)row1 * res_ld + cg];
        }
        if (RELU) { v0 = fmaxf(v0, 0.f); v1 = fmaxf(v1, 0.f); }
        OUT[(size_t)row0 * out_ld + out_coloff + cg] = v0;
        OUT[(size_t)row1 * out_ld + out_coloff + cg] = v1;
    }
}

// ---------------------------------------------------------------------------
// scores[b,l] = sqrt(C) * (U[b] . lfb[b,l])  -- one wave per (b,l) row,
// float4 streaming loads (global_load_b128), HBM-bound pass #1 over lfb.
// ---------------------------------------------------------------------------
__global__ void scores_dot(const float* __restrict__ lfb,
                           const float* __restrict__ U,
                           float* __restrict__ scores)
{
    const int lane = threadIdx.x & 31;
    const int wave = blockIdx.x * (blockDim.x >> 5) + (threadIdx.x >> 5);
    const int b = wave >> 10;
    const int l = wave & (L_DIM - 1);
    const float4* __restrict__ row =
        (const float4*)(lfb + ((size_t)b * L_DIM + l) * C_DIM);
    const float4* __restrict__ u4 = (const float4*)(U + (size_t)b * C_DIM);
    float acc = 0.f;
    #pragma unroll
    for (int i = 0; i < C_DIM / 4 / 32; ++i) {      // 12 x float4 per lane
        float4 a = row[lane + 32 * i];
        float4 u = u4[lane + 32 * i];
        acc += a.x * u.x + a.y * u.y + a.z * u.z + a.w * u.w;
    }
    #pragma unroll
    for (int off = 16; off > 0; off >>= 1)
        acc += __shfl_down(acc, off, 32);
    if (lane == 0) scores[wave] = acc * sqrtf((float)C_DIM);
}

// ---------------------------------------------------------------------------
// In-place softmax over L per batch; also zeros V[b] for the atomic pass.
// ---------------------------------------------------------------------------
__global__ void softmax_zeroV(float* __restrict__ scores, float* __restrict__ V)
{
    __shared__ float red[32];
    const int b = blockIdx.x;
    const int t = threadIdx.x;                       // 0..1023
    float s = scores[b * L_DIM + t];

    float m = s;
    for (int off = 16; off > 0; off >>= 1) m = fmaxf(m, __shfl_down(m, off, 32));
    if ((t & 31) == 0) red[t >> 5] = m;
    __syncthreads();
    if (t < 32) {
        float v = red[t];
        for (int off = 16; off > 0; off >>= 1) v = fmaxf(v, __shfl_down(v, off, 32));
        if (t == 0) red[0] = v;
    }
    __syncthreads();
    m = red[0];
    __syncthreads();

    float e = __expf(s - m);
    float sum = e;
    for (int off = 16; off > 0; off >>= 1) sum += __shfl_down(sum, off, 32);
    if ((t & 31) == 0) red[t >> 5] = sum;
    __syncthreads();
    if (t < 32) {
        float v = red[t];
        for (int off = 16; off > 0; off >>= 1) v += __shfl_down(v, off, 32);
        if (t == 0) red[0] = v;
    }
    __syncthreads();
    scores[b * L_DIM + t] = e / red[0];

    for (int c = t; c < C_DIM; c += blockDim.x) V[(size_t)b * C_DIM + c] = 0.f;
}

// ---------------------------------------------------------------------------
// V[b] += sum_l attn[b,l] * lfb[b,l,:]  -- HBM-bound pass #2 over lfb.
// Grid (16 L-chunks x 32 batches), 384 threads cover all 1536 columns as
// float4; partials merged with global_atomic_add_f32.  A software prefetch
// (global_prefetch_b8) keeps the stream ahead of the FMA chain.
// ---------------------------------------------------------------------------
__global__ void v_accum(const float* __restrict__ lfb,
                        const float* __restrict__ attn,
                        float* __restrict__ V)
{
    const int b  = blockIdx.y;
    const int lc = blockIdx.x;          // 0..15
    const int t  = threadIdx.x;         // 0..383 -> float4 column index
    float4 acc = {0.f, 0.f, 0.f, 0.f};
    const float4* __restrict__ base =
        (const float4*)(lfb + (size_t)b * L_DIM * C_DIM);
    const int l0 = lc * (L_DIM / 16);
    for (int l = l0; l < l0 + L_DIM / 16; ++l) {
        const float a = attn[b * L_DIM + l];
        float4 v = base[(size_t)l * (C_DIM / 4) + t];
        __builtin_prefetch(&base[(size_t)(l + 4) * (C_DIM / 4) + t], 0, 1);
        acc.x += a * v.x; acc.y += a * v.y; acc.z += a * v.z; acc.w += a * v.w;
    }
    float* out = V + (size_t)b * C_DIM + t * 4;
    atomicAdd(out + 0, acc.x); atomicAdd(out + 1, acc.y);
    atomicAdd(out + 2, acc.z); atomicAdd(out + 3, acc.w);
}

// ---------------------------------------------------------------------------
// Oln = relu(layernorm(O)); also feats[:, :C] = x.
// ---------------------------------------------------------------------------
__global__ void ln_relu_copyx(const float* __restrict__ O,
                              const float* __restrict__ g,
                              const float* __restrict__ beta,
                              const float* __restrict__ x,
                              float* __restrict__ Oln,
                              float* __restrict__ feats)
{
    __shared__ float red[8];
    const int b = blockIdx.x;
    const int t = threadIdx.x;          // 256 threads, 6 elems each
    float v[6];
    float s = 0.f;
    #pragma unroll
    for (int i = 0; i < 6; ++i) { v[i] = O[(size_t)b * C_DIM + t + 256 * i]; s += v[i]; }
    for (int off = 16; off > 0; off >>= 1) s += __shfl_down(s, off, 32);
    if ((t & 31) == 0) red[t >> 5] = s;
    __syncthreads();
    if (t == 0) { float tot = 0.f; for (int i = 0; i < 8; ++i) tot += red[i]; red[0] = tot; }
    __syncthreads();
    const float mu = red[0] / (float)C_DIM;
    __syncthreads();
    float vs = 0.f;
    #pragma unroll
    for (int i = 0; i < 6; ++i) { float d = v[i] - mu; vs += d * d; }
    for (int off = 16; off > 0; off >>= 1) vs += __shfl_down(vs, off, 32);
    if ((t & 31) == 0) red[t >> 5] = vs;
    __syncthreads();
    if (t == 0) { float tot = 0.f; for (int i = 0; i < 8; ++i) tot += red[i]; red[0] = tot; }
    __syncthreads();
    const float rstd = rsqrtf(red[0] / (float)C_DIM + 1e-5f);
    #pragma unroll
    for (int i = 0; i < 6; ++i) {
        const int c = t + 256 * i;
        float y = (v[i] - mu) * rstd * g[c] + beta[c];
        Oln[(size_t)b * C_DIM + c]    = fmaxf(y, 0.f);
        feats[(size_t)b * D2_DIM + c] = x[(size_t)b * C_DIM + c];
    }
}

// ---------------------------------------------------------------------------
// Four scalar heads: p1/p2/p3 over 1024-slices, pt over the full 3072.
// ---------------------------------------------------------------------------
__global__ void heads(const float* __restrict__ h,
                      const float* fc1_w, const float* fc1_b,
                      const float* fc2_w, const float* fc2_b,
                      const float* fc3_w, const float* fc3_b,
                      const float* fct_w, const float* fct_b,
                      float* __restrict__ out)
{
    __shared__ float red[4][8];
    const int b = blockIdx.x;
    const int t = threadIdx.x;          // 256
    float p[4] = {0.f, 0.f, 0.f, 0.f};
    for (int i = t; i < D2_DIM; i += 256) {
        const float f = h[(size_t)b * D2_DIM + i];
        p[3] += f * fct_w[i];
        if (i < 1024)       p[0] += f * fc1_w[i];
        else if (i < 2048)  p[1] += f * fc2_w[i - 1024];
        else                p[2] += f * fc3_w[i - 2048];
    }
    #pragma unroll
    for (int j = 0; j < 4; ++j) {
        float v = p[j];
        for (int off = 16; off > 0; off >>= 1) v += __shfl_down(v, off, 32);
        if ((t & 31) == 0) red[j][t >> 5] = v;
    }
    __syncthreads();
    if (t == 0) {
        float tot[4] = {0.f, 0.f, 0.f, 0.f};
        for (int w = 0; w < 8; ++w)
            for (int j = 0; j < 4; ++j) tot[j] += red[j][w];
        out[b * 4 + 0] = tot[0] + fc1_b[0];
        out[b * 4 + 1] = tot[1] + fc2_b[0];
        out[b * 4 + 2] = tot[2] + fc3_b[0];
        out[b * 4 + 3] = tot[3] + fct_b[0];
    }
}

// ---------------------------------------------------------------------------
extern "C" void kernel_launch(void* const* d_in, const int* in_sizes, int n_in,
                              void* d_out, int out_size, void* d_ws, size_t ws_size,
                              hipStream_t stream)
{
    (void)in_sizes; (void)n_in; (void)out_size; (void)ws_size;
    const float* x        = (const float*)d_in[0];
    const float* lfb      = (const float*)d_in[1];
    const float* theta_w  = (const float*)d_in[2];
    const float* theta_b  = (const float*)d_in[3];
    const float* phi_w    = (const float*)d_in[4];
    /* phi_b (d_in[5]) adds a per-batch constant to all scores -> cancels in softmax */
    const float* gi_w     = (const float*)d_in[6];
    const float* gi_b     = (const float*)d_in[7];
    const float* ln_g     = (const float*)d_in[8];
    const float* ln_b     = (const float*)d_in[9];
    const float* fc_w     = (const float*)d_in[10];
    const float* fc_b     = (const float*)d_in[11];
    const float* fc_nlb_w = (const float*)d_in[12];
    const float* fc_nlb_b = (const float*)d_in[13];
    const float* fc1_w    = (const float*)d_in[14];
    const float* fc1_b    = (const float*)d_in[15];
    const float* fc2_w    = (const float*)d_in[16];
    const float* fc2_b    = (const float*)d_in[17];
    const float* fc3_w    = (const float*)d_in[18];
    const float* fc3_b    = (const float*)d_in[19];
    const float* fct_w    = (const float*)d_in[20];
    const float* fct_b    = (const float*)d_in[21];

    float* ws     = (float*)d_ws;
    float* theta  = ws;                         // 32*1536
    float* U      = theta  + B_DIM * C_DIM;     // 32*1536
    float* scores = U      + B_DIM * C_DIM;     // 32*1024 (becomes attn)
    float* V      = scores + B_DIM * L_DIM;     // 32*1536
    float* O      = V      + B_DIM * C_DIM;     // 32*1536
    float* Oln    = O      + B_DIM * C_DIM;     // 32*1536
    float* feats  = Oln    + B_DIM * C_DIM;     // 32*3072
    float* hbuf   = feats  + B_DIM * D2_DIM;    // 32*3072

    // theta = x @ theta_w^T + theta_b
    gemm32_bf16_wmma<true, false, true, false><<<C_DIM / 16, 32, 0, stream>>>(
        x, C_DIM, theta_w, theta_b, nullptr, 0, theta, C_DIM, C_DIM, 0);
    // U = theta @ phi_w   (folds the phi projection into the query)
    gemm32_bf16_wmma<false, false, false, false><<<C_DIM / 16, 32, 0, stream>>>(
        theta, C_DIM, phi_w, nullptr, nullptr, 0, U, C_DIM, C_DIM, 0);
    // scores (streaming pass 1 over lfb)
    scores_dot<<<(B_DIM * L_DIM) / 8, 256, 0, stream>>>(lfb, U, scores);
    // softmax + zero V
    softmax_zeroV<<<B_DIM, L_DIM, 0, stream>>>(scores, V);
    // V = attn-weighted sum of lfb rows (streaming pass 2 over lfb)
    v_accum<<<dim3(16, B_DIM), C_DIM / 4, 0, stream>>>(lfb, scores, V);
    // O = V @ gi_w^T + gi_b
    gemm32_bf16_wmma<true, false, true, false><<<C_DIM / 16, 32, 0, stream>>>(
        V, C_DIM, gi_w, gi_b, nullptr, 0, O, C_DIM, C_DIM, 0);
    // Oln = relu(LN(O)); feats[:, :C] = x
    ln_relu_copyx<<<B_DIM, 256, 0, stream>>>(O, ln_g, ln_b, x, Oln, feats);
    // feats[:, C:] = Oln @ fc_w^T + fc_b + x   (residual)
    gemm32_bf16_wmma<true, false, true, true><<<C_DIM / 16, 32, 0, stream>>>(
        Oln, C_DIM, fc_w, fc_b, x, C_DIM, feats, C_DIM, D2_DIM, C_DIM);
    // h = relu(feats @ fc_nlb_w^T + fc_nlb_b)
    gemm32_bf16_wmma<true, true, true, false><<<D2_DIM / 16, 32, 0, stream>>>(
        feats, D2_DIM, fc_nlb_w, fc_nlb_b, nullptr, 0, hbuf, D2_DIM, D2_DIM, 0);
    // four heads -> (32,4)
    heads<<<B_DIM, 256, 0, stream>>>(hbuf, fc1_w, fc1_b, fc2_w, fc2_b,
                                     fc3_w, fc3_b, fct_w, fct_b, (float*)d_out);
}
